// LISTA_Net_69174743269838
// MI455X (gfx1250) — matrix operand order repaired and checked by reading, not compiled, not run on hardware
//
#include <hip/hip_runtime.h>
#include <math.h>

typedef __bf16 bf16;
typedef __attribute__((ext_vector_type(16))) __bf16 v16bf;
typedef __attribute__((ext_vector_type(8)))  __bf16 v8bf;
typedef __attribute__((ext_vector_type(8)))  float  v8f;

#define NDIM 2048

__device__ __forceinline__ v8bf cvt8(float4 f0, float4 f1) {
  v8bf r;
  r[0] = (bf16)f0.x; r[1] = (bf16)f0.y; r[2] = (bf16)f0.z; r[3] = (bf16)f0.w;
  r[4] = (bf16)f1.x; r[5] = (bf16)f1.y; r[6] = (bf16)f1.z; r[7] = (bf16)f1.w;
  return r;
}

__device__ __forceinline__ v16bf frag32(const bf16* p) {
  const v8bf lo = *(const v8bf*)p;
  const v8bf hi = *(const v8bf*)(p + 8);
  v16bf f;
  #pragma unroll
  for (int e = 0; e < 8; ++e) { f[e] = lo[e]; f[e + 8] = hi[e]; }
  return f;
}

// async global->LDS copy of 32 contiguous bytes per lane (2 x b128).
// ldsaddr: workgroup-relative LDS byte offset (low 32 bits of flat pointer);
// the instruction adds the same INST_OFFSET to both LDS and global sides.
__device__ __forceinline__ void async_copy_b256(unsigned ldsaddr, const void* gptr) {
  unsigned long long ga = (unsigned long long)gptr;
  asm volatile(
      "global_load_async_to_lds_b128 %0, %1, off\n\t"
      "global_load_async_to_lds_b128 %0, %1, off offset:16"
      :: "v"(ldsaddr), "v"(ga) : "memory");
}

__device__ __forceinline__ void async_wait0() {
  asm volatile("s_wait_asynccnt 0" ::: "memory");
}

// ---- vectorized f32 -> bf16 copy (count % 8 == 0) ---------------------------
__global__ void conv_bf16_kernel(const float* __restrict__ src, bf16* __restrict__ dst,
                                 int count) {
  int o = (blockIdx.x * blockDim.x + threadIdx.x) * 8;
  if (o >= count) return;
  float4 f0 = *(const float4*)(src + o);
  float4 f1 = *(const float4*)(src + o + 4);
  *(v8bf*)(dst + o) = cvt8(f0, f1);
}

// ---- bf16 WMMA GEMM: C(rows x NDIM) = A(rows x K, f32) * Bt^T ---------------
// Bt is N-major bf16: Bt[n][k]  (W1 is already N-major; W2 by symmetry)
// Block tile 128x128, 8 waves (4Mx2N), wave tile 32x64 = 8 WMMA / K-step.
// B tiles staged with GLOBAL_LOAD_ASYNC_TO_LDS_B128 (ASYNCcnt); A tiles go
// through registers for the f32->bf16 conversion.
// mode 1: wyOut = acc, xOut = mu[0]*acc     mode 0: uOut = acc
__global__ void __launch_bounds__(256) gemm_wmma_kernel(
    const float* __restrict__ A,
    const bf16*  __restrict__ Bt,
    int K,
    float* __restrict__ uOut,
    float* __restrict__ wyOut,
    float* __restrict__ xOut,
    const float* __restrict__ mu,
    int mode)
{
  // Row stride 40 bf16 = 80 B: 16B-aligned b128 accesses, conflict-free
  // fragment reads. A-tile k stored in natural order, so each lane's A
  // fragment is one contiguous 32-byte run.
  __shared__ bf16 As[2][128][40];
  __shared__ bf16 Bs[2][128][40];

  const int tid   = threadIdx.x;
  const int lane  = tid & 31;
  const int wave  = tid >> 5;       // 0..7
  const int waveM = wave >> 1;      // 0..3 (32-row slabs)
  const int waveN = wave & 1;       // 0..1 (64-col slabs)
  const int half  = lane >> 4;
  const int lr    = lane & 15;

  const int rowBase = blockIdx.y * 128;
  const int colBase = blockIdx.x * 128;

  // staging: A 128x32 f32 (16 floats/thread), B 128x32 bf16 (16 bf16/thread)
  const int ar  = tid >> 1;                 // 0..127
  const int ah  = tid & 1;                  // k-half: groups {0,1} or {2,3}
  const int ac  = ah << 4;                  // source col (16 floats)
  const int apc0 = (ah == 0) ? 0  : 16;     // dest cols (contiguous k)
  const int apc1 = (ah == 0) ? 8  : 24;
  const int bn  = tid >> 1;                 // 0..127
  const int bc  = (tid & 1) << 4;

  const float* Aptr = A  + (size_t)(rowBase + ar) * K + ac;
  const bf16*  Bptr = Bt + (size_t)(colBase + bn) * K + bc;

  // workgroup-relative LDS offsets for the async B staging (per buffer)
  const unsigned bs_lds0 = (unsigned)(size_t)&Bs[0][bn][bc];
  const unsigned bs_lds1 = (unsigned)(size_t)&Bs[1][bn][bc];

  v8f acc[8] = {{}, {}, {}, {}, {}, {}, {}, {}};
  const int nk = K >> 5;

  // prologue: stage k-block 0 into buffer 0
  async_copy_b256(bs_lds0, Bptr);
  float4 fa0 = *(const float4*)(Aptr + 0);
  float4 fa1 = *(const float4*)(Aptr + 4);
  float4 fa2 = *(const float4*)(Aptr + 8);
  float4 fa3 = *(const float4*)(Aptr + 12);
  *(v8bf*)&As[0][ar][apc0] = cvt8(fa0, fa1);
  *(v8bf*)&As[0][ar][apc1] = cvt8(fa2, fa3);
  async_wait0();

  for (int kb = 0; kb < nk; ++kb) {
    __syncthreads();
    const int buf = kb & 1;
    const bool more = (kb + 1 < nk);

    // issue next k-block transfers (overlap with WMMA below)
    if (more) {
      const int koff = (kb + 1) << 5;
      async_copy_b256((buf == 0) ? bs_lds1 : bs_lds0, Bptr + koff);
      fa0 = *(const float4*)(Aptr + koff);
      fa1 = *(const float4*)(Aptr + koff + 4);
      fa2 = *(const float4*)(Aptr + koff + 8);
      fa3 = *(const float4*)(Aptr + koff + 12);
    }
    // prefetch two k-blocks ahead (global_prefetch_b8)
    if (kb + 2 < nk) {
      const int kpf = (kb + 2) << 5;
      __builtin_prefetch(Aptr + kpf, 0, 3);
      __builtin_prefetch(Bptr + kpf, 0, 3);
    }

    // fragments: 2 A tiles, 4 B tiles
    const v16bf a0 = frag32(&As[buf][waveM * 32 + lr][half * 16]);
    const v16bf a1 = frag32(&As[buf][waveM * 32 + 16 + lr][half * 16]);
    v16bf bf[4];
    #pragma unroll
    for (int j = 0; j < 4; ++j)
      bf[j] = frag32(&Bs[buf][waveN * 64 + j * 16 + lr][half * 16]);

    #pragma unroll
    for (int j = 0; j < 4; ++j) {
      acc[j]     = __builtin_amdgcn_wmma_f32_16x16x32_bf16(
                       false, a0, false, bf[j], (short)0, acc[j],     false, false);
      acc[4 + j] = __builtin_amdgcn_wmma_f32_16x16x32_bf16(
                       false, a1, false, bf[j], (short)0, acc[4 + j], false, false);
    }

    // convert + store next A tile; wait for our async B before publishing
    if (more) {
      const int nbuf = buf ^ 1;
      *(v8bf*)&As[nbuf][ar][apc0] = cvt8(fa0, fa1);
      *(v8bf*)&As[nbuf][ar][apc1] = cvt8(fa2, fa3);
      async_wait0();
    }
  }

  // epilogue: D element v -> row = tileRow + v + 8*half, col = tileCol + lr
  const int row0 = rowBase + waveM * 32 + 8 * half;
  if (mode == 1) {
    const float mu0 = mu[0];
    #pragma unroll
    for (int h = 0; h < 2; ++h) {
      #pragma unroll
      for (int j = 0; j < 4; ++j) {
        const int col = colBase + waveN * 64 + j * 16 + lr;
        #pragma unroll
        for (int v = 0; v < 8; ++v) {
          size_t i0 = (size_t)(row0 + h * 16 + v) * NDIM + col;
          float c = acc[h * 4 + j][v];
          wyOut[i0] = c;
          xOut[i0]  = mu0 * c;
        }
      }
    }
  } else {
    #pragma unroll
    for (int h = 0; h < 2; ++h) {
      #pragma unroll
      for (int j = 0; j < 4; ++j) {
        const int col = colBase + waveN * 64 + j * 16 + lr;
        #pragma unroll
        for (int v = 0; v < 8; ++v) {
          uOut[(size_t)(row0 + h * 16 + v) * NDIM + col] = acc[h * 4 + j][v];
        }
      }
    }
  }
}

// ---- fused update + exact top-k soft-threshold per row ----------------------
__global__ void __launch_bounds__(256) shrink_kernel(
    const float* __restrict__ u,
    const float* __restrict__ wy,
    float* __restrict__ x,
    const float* __restrict__ mu,
    const float* __restrict__ beta,
    int t, int kSel)
{
  __shared__ __align__(16) float vrow[NDIM];
  __shared__ unsigned hist[256];
  __shared__ unsigned s_prefix, s_kk;

  const int tid = threadIdx.x;
  const int row = blockIdx.x;
  const float mu_t   = mu[t];
  const float beta_t = beta[t];
  const size_t base  = (size_t)row * NDIM;

  const float4* u4  = (const float4*)(u + base);
  const float4* wy4 = (const float4*)(wy + base);
  float4*       x4  = (float4*)(x + base);

  for (int i = tid; i < NDIM / 4; i += 256) {
    float4 xv = x4[i], uv = u4[i], wv = wy4[i];
    float4 v;
    v.x = xv.x - mu_t * (uv.x - wv.x);
    v.y = xv.y - mu_t * (uv.y - wv.y);
    v.z = xv.z - mu_t * (uv.z - wv.z);
    v.w = xv.w - mu_t * (uv.w - wv.w);
    *(float4*)&vrow[i * 4] = v;
  }
  if (tid == 0) { s_prefix = 0u; s_kk = (unsigned)kSel; }
  __syncthreads();

  // 4-pass radix select for the kSel-th largest |v| (abs f32 bits are monotone)
  for (int shift = 24; shift >= 0; shift -= 8) {
    hist[tid] = 0u;
    __syncthreads();
    const unsigned pre = s_prefix;
    for (int i = tid; i < NDIM; i += 256) {
      unsigned b = __float_as_uint(fabsf(vrow[i]));
      bool match = (shift == 24) || ((b >> (shift + 8)) == (pre >> (shift + 8)));
      if (match) atomicAdd(&hist[(b >> shift) & 255u], 1u);
    }
    __syncthreads();
    if (tid == 0) {
      unsigned kk = s_kk, accu = 0u;
      for (int d = 255; d >= 0; --d) {
        unsigned c = hist[d];
        if (accu + c >= kk) { s_kk = kk - accu; s_prefix = pre | ((unsigned)d << shift); break; }
        accu += c;
      }
    }
    __syncthreads();
  }

  const float thr = __uint_as_float(s_prefix);
  for (int i = tid; i < NDIM / 4; i += 256) {
    float4 v = *(float4*)&vrow[i * 4];
    float4 r;
    {
      float a0 = fabsf(v.x); r.x = (a0 >= thr) ? v.x : copysignf(fmaxf(a0 - beta_t, 0.f), v.x);
      float a1 = fabsf(v.y); r.y = (a1 >= thr) ? v.y : copysignf(fmaxf(a1 - beta_t, 0.f), v.y);
      float a2 = fabsf(v.z); r.z = (a2 >= thr) ? v.z : copysignf(fmaxf(a2 - beta_t, 0.f), v.z);
      float a3 = fabsf(v.w); r.w = (a3 >= thr) ? v.w : copysignf(fmaxf(a3 - beta_t, 0.f), v.w);
    }
    x4[i] = r;
  }
}

// ---- launcher ---------------------------------------------------------------
extern "C" void kernel_launch(void* const* d_in, const int* in_sizes, int n_in,
                              void* d_out, int out_size, void* d_ws, size_t ws_size,
                              hipStream_t stream) {
  (void)in_sizes; (void)n_in; (void)out_size; (void)ws_size;

  const float* y    = (const float*)d_in[0];   // (8192, 512)
  const float* W1   = (const float*)d_in[1];   // (2048, 512)  == Bt for GEMM1
  const float* W2   = (const float*)d_in[2];   // (2048, 2048) symmetric == Bt for GEMM2
  const float* mu   = (const float*)d_in[3];   // (6,)
  const float* beta = (const float*)d_in[4];   // (6,)
  float* x = (float*)d_out;                    // (8192, 2048)

  const int Bn = 8192, M = 512, N = 2048, T = 5;

  char* ws = (char*)d_ws;
  bf16*  w1b = (bf16*)ws;                                         // 2 MB  (N x M)
  bf16*  w2b = (bf16*)(ws + (size_t)N * M * 2);                   // 8 MB  (N x N)
  float* wy  = (float*)(ws + (size_t)N * M * 2 + (size_t)N * N * 2);   // 64 MB
  float* u   = (float*)((char*)wy + (size_t)Bn * N * 4);               // 64 MB

  conv_bf16_kernel<<<(N * M / 8 + 255) / 256, 256, 0, stream>>>(W1, w1b, N * M);
  conv_bf16_kernel<<<(N * N / 8 + 255) / 256, 256, 0, stream>>>(W2, w2b, N * N);

  dim3 grid(N / 128, Bn / 128);   // 16 x 64 blocks
  // Wy = y @ W1^T ; x0 = mu0 * Wy  (t=0 shrink is identity: k=0)
  gemm_wmma_kernel<<<grid, 256, 0, stream>>>(y, w1b, M, nullptr, wy, x, mu, 1);

  for (int t = 1; t <= T; ++t) {
    // u = x @ W2 (W2 symmetric)
    gemm_wmma_kernel<<<grid, 256, 0, stream>>>(x, w2b, N, u, nullptr, nullptr, mu, 0);
    double frac = 0.012 * t; if (frac > 0.12) frac = 0.12;
    int kSel = (int)(frac * N);                 // 24,49,73,98,122
    shrink_kernel<<<Bn, 256, 0, stream>>>(u, wy, x, mu, beta, t, kSel);
  }
}
